// BasketballGNN_46583215292449
// MI455X (gfx1250) — matrix operand order repaired and checked by reading, not compiled, not run on hardware
//
#include <hip/hip_runtime.h>
#include <hip/hip_bf16.h>

// ---------------- problem constants ----------------
#define NN 50000
#define NE 1600000
#define IN_DIM 6
#define HID 64
#define OUTD 32

typedef __attribute__((ext_vector_type(16))) _Float16 v16h;
typedef __attribute__((ext_vector_type(8)))  float    v8f;

union F8 { v8f v; float f[8]; };

// ---------------- workspace layout (bytes) ----------------
// h16   : NN*64 f16                      @ 0
// sums  : NN*64 f32                      @ 6,400,000
// counts: NN    f32                      @ 19,200,000
// packed weights (f16 frags, 512 halves each) @ 19,400,000
#define WS_H16   0
#define WS_SUMS  6400000
#define WS_CNT   19200000
#define WS_W     19400000
// frag-unit offsets inside packed weight arena
#define F_ENC1 0    // 4 frags  (K=6->1 ktile, 4 ntiles)
#define F_ENC2 4    // 8 frags  (K=64, N=64)
#define F_MSG1 12   // 16 frags (K=128,N=64)
#define F_MSG2 28   // 8 frags
#define F_UPD1 36   // 16 frags
#define F_UPD2 52   // 4 frags  (K=64,N=32)
#define F_TAC1 56   // 4 frags  (K=32,N=64)
#define F_TAC2 60   // 2 frags  (K=64,N=16)

// ---------------- WMMA helpers ----------------
__device__ __forceinline__ v8f wmma32(v16h a, v16h b, v8f c) {
  return __builtin_amdgcn_wmma_f32_16x16x32_f16(
      false, a, false, b, (short)0, c, false, false);
}

// load a pre-packed B fragment (32 contiguous bytes per lane)
__device__ __forceinline__ v16h ld_bfrag(const _Float16* base, int frag, int lane) {
  const uint4* p = (const uint4*)(base + (size_t)frag * 512 + lane * 16);
  union { v16h h; uint4 q[2]; } u;
  u.q[0] = p[0];
  u.q[1] = p[1];
  return u.h;
}

// A fragment (16x32 f16) from a row-major 16-row tile with leading dim `ldh` halves.
// lane 0-15: row=lane, K = k0+[0..7], k0+[16..23]; lane 16-31: row=lane-16, +8.
__device__ __forceinline__ v16h ld_afrag(const _Float16* tile, int ldh, int k0, int lane) {
  int r = lane & 15;
  int koff = k0 + ((lane >> 4) << 3);
  const _Float16* p = tile + r * ldh + koff;
  union { v16h h; uint4 q[2]; } u;
  u.q[0] = *(const uint4*)(p);
  u.q[1] = *(const uint4*)(p + 16);
  return u.h;
}

// A fragment where each lane has its own 64-half node vector (edge gather case)
__device__ __forceinline__ v16h ld_afrag_vec(const _Float16* vec, int k0, int lane) {
  int koff = k0 + ((lane >> 4) << 3);
  union { v16h h; uint4 q[2]; } u;
  u.q[0] = *(const uint4*)(vec + koff);
  u.q[1] = *(const uint4*)(vec + koff + 16);
  return u.h;
}

// ---------------- utility kernels ----------------
__global__ void gnn_zero_kernel(float* __restrict__ p, long long n) {
  long long i = (long long)blockIdx.x * blockDim.x + threadIdx.x;
  if (i < n) p[i] = 0.0f;
}

// Pack fp32 W[K,N] (row-major, x@W convention) into per-fragment f16 layout.
// frag f = kt*ntiles + nt ; within frag lane holds 16 contiguous halves:
//   col n = nt*16 + (lane&15); K = kt*32 + (j<8 ? 8*(lane>>4)+j : 16+8*(lane>>4)+(j-8)), j=0..15
__global__ __launch_bounds__(512)
void gnn_pack_weights(const float* __restrict__ W, int K, int Nn, int ntiles,
                      _Float16* __restrict__ dst) {
  int f = blockIdx.x;
  int t = threadIdx.x;       // 0..511
  int lane = t >> 4;         // 0..31
  int j = t & 15;            // 0..15
  int kt = f / ntiles, nt = f % ntiles;
  int n = nt * 16 + (lane & 15);
  int k = kt * 32 + ((j < 8) ? ((lane >> 4) << 3) + j : 16 + ((lane >> 4) << 3) + (j - 8));
  float v = (k < K && n < Nn) ? W[(size_t)k * Nn + n] : 0.0f;
  dst[(size_t)f * 512 + lane * 16 + j] = (_Float16)v;
}

__global__ void gnn_count_kernel(const int* __restrict__ col, float* __restrict__ counts, int E) {
  int e = blockIdx.x * blockDim.x + threadIdx.x;
  if (e < E) atomicAdd(&counts[col[e]], 1.0f);
}

// ---------------- node encoder: h = relu(x@W1+b1)@W2+b2 ----------------
__global__ __launch_bounds__(256)
void gnn_encode_kernel(const float* __restrict__ feat,
                       const _Float16* __restrict__ w1, const float* __restrict__ b1,
                       const _Float16* __restrict__ w2, const float* __restrict__ b2,
                       _Float16* __restrict__ h16, int N) {
  __shared__ __align__(16) _Float16 smem[8][16 * 64];
  int lane = threadIdx.x & 31;
  int wave = threadIdx.x >> 5;
  _Float16* tile = smem[wave];
  int nb = blockIdx.x * 128 + wave * 16;
  int r = lane & 15;
  int n0 = r;
  int node = nb + r; if (node > N - 1) node = N - 1;

  // stage x padded to K=32 (cols >= IN_DIM are zero)
  int cbase = (lane >> 4) * 16;
  for (int c = 0; c < 16; ++c) {
    int col = cbase + c;
    float v = (col < IN_DIM) ? feat[(size_t)node * IN_DIM + col] : 0.0f;
    tile[r * 32 + col] = (_Float16)v;
  }
  __syncthreads();

  v16h a = ld_afrag(tile, 32, 0, lane);
  F8 acc[4];
  for (int nt = 0; nt < 4; ++nt) {
    float b = b1[nt * 16 + n0];
    for (int i = 0; i < 8; ++i) acc[nt].f[i] = b;
    acc[nt].v = wmma32(a, ld_bfrag(w1, nt, lane), acc[nt].v);
  }
  __syncthreads();
  for (int nt = 0; nt < 4; ++nt)
    for (int i = 0; i < 8; ++i) {
      float v = acc[nt].f[i]; v = v > 0.0f ? v : 0.0f;   // ReLU
      int m = (lane < 16) ? i : i + 8;
      tile[m * 64 + nt * 16 + n0] = (_Float16)v;
    }
  __syncthreads();

  v16h a0 = ld_afrag(tile, 64, 0, lane);
  v16h a1 = ld_afrag(tile, 64, 32, lane);
  for (int nt = 0; nt < 4; ++nt) {
    F8 o; float b = b2[nt * 16 + n0];
    for (int i = 0; i < 8; ++i) o.f[i] = b;
    o.v = wmma32(a0, ld_bfrag(w2, 0 * 4 + nt, lane), o.v);
    o.v = wmma32(a1, ld_bfrag(w2, 1 * 4 + nt, lane), o.v);
    for (int i = 0; i < 8; ++i) {
      int m = (lane < 16) ? i : i + 8;
      int nd = nb + m;
      if (nd < N) h16[(size_t)nd * 64 + nt * 16 + n0] = (_Float16)o.f[i];
    }
  }
}

// ---------------- edge messages + scatter-add ----------------
__global__ __launch_bounds__(256)
void gnn_edge_kernel(const int* __restrict__ erow, const int* __restrict__ ecol,
                     const _Float16* __restrict__ h16,
                     const _Float16* __restrict__ w1, const float* __restrict__ b1,
                     const _Float16* __restrict__ w2, const float* __restrict__ b2,
                     float* __restrict__ sums, int E) {
  __shared__ __align__(16) _Float16 smem[8][16 * 64];
  int lane = threadIdx.x & 31;
  int wave = threadIdx.x >> 5;
  _Float16* tile = smem[wave];
  int eb = blockIdx.x * 128 + wave * 16;
  int r = lane & 15;
  int n0 = r;
  int e = eb + r; if (e > E - 1) e = E - 1;
  int nr = erow[e];
  int nc = ecol[e];

  // concat(h[row], h[col]) lives directly in fragment space: K 0..63 from row, 64..127 from col
  v16h a[4];
  a[0] = ld_afrag_vec(h16 + (size_t)nr * 64, 0, lane);
  a[1] = ld_afrag_vec(h16 + (size_t)nr * 64, 32, lane);
  a[2] = ld_afrag_vec(h16 + (size_t)nc * 64, 0, lane);
  a[3] = ld_afrag_vec(h16 + (size_t)nc * 64, 32, lane);

  for (int nt = 0; nt < 4; ++nt) {
    F8 acc; float b = b1[nt * 16 + n0];
    for (int i = 0; i < 8; ++i) acc.f[i] = b;
    for (int kc = 0; kc < 4; ++kc)
      acc.v = wmma32(a[kc], ld_bfrag(w1, kc * 4 + nt, lane), acc.v);
    for (int i = 0; i < 8; ++i) {
      float v = acc.f[i]; v = v > 0.0f ? v : 0.0f;   // ReLU
      int m = (lane < 16) ? i : i + 8;
      tile[m * 64 + nt * 16 + n0] = (_Float16)v;
    }
  }
  __syncthreads();

  v16h m0 = ld_afrag(tile, 64, 0, lane);
  v16h m1 = ld_afrag(tile, 64, 32, lane);
  for (int nt = 0; nt < 4; ++nt) {
    F8 o; float b = b2[nt * 16 + n0];
    for (int i = 0; i < 8; ++i) o.f[i] = b;
    o.v = wmma32(m0, ld_bfrag(w2, 0 * 4 + nt, lane), o.v);
    o.v = wmma32(m1, ld_bfrag(w2, 1 * 4 + nt, lane), o.v);
    for (int i = 0; i < 8; ++i) {
      int m = (lane < 16) ? i : i + 8;
      int e2 = eb + m;
      if (e2 < E) {
        int d = ecol[e2];
        atomicAdd(&sums[(size_t)d * 64 + nt * 16 + n0], o.f[i]);
      }
    }
  }
}

// ---------------- update MLP + tactical head ----------------
__global__ __launch_bounds__(256)
void gnn_update_kernel(const _Float16* __restrict__ h16,
                       const float* __restrict__ sums, const float* __restrict__ counts,
                       const _Float16* __restrict__ wu1, const float* __restrict__ bu1,
                       const _Float16* __restrict__ wu2, const float* __restrict__ bu2,
                       const _Float16* __restrict__ wt1, const float* __restrict__ bt1,
                       const _Float16* __restrict__ wt2, const float* __restrict__ bt2,
                       const float* __restrict__ Wt3, const float* __restrict__ bt3,
                       float* __restrict__ out, int N) {
  __shared__ __align__(16) _Float16 smem[8][16 * 128];
  __shared__ float t2s[8][16 * 16];
  int lane = threadIdx.x & 31;
  int wave = threadIdx.x >> 5;
  _Float16* tile = smem[wave];
  int nb = blockIdx.x * 128 + wave * 16;
  int r = lane & 15;
  int n0 = r;
  int node = nb + r; if (node > N - 1) node = N - 1;

  // stage concat(h, mean_agg) as 16x128 f16 tile
  if (lane < 16) {
    const uint4* src = (const uint4*)(h16 + (size_t)node * 64);
    uint4* dst = (uint4*)(tile + r * 128);
    for (int q = 0; q < 8; ++q) dst[q] = src[q];
  } else {
    float cnt = counts[node];
    float inv = cnt > 0.0f ? 1.0f / cnt : 0.0f;
    for (int j = 0; j < 64; ++j) {
      float s = sums[(size_t)node * 64 + j] * inv;
      tile[r * 128 + 64 + j] = (_Float16)s;
    }
  }
  __syncthreads();
  v16h a0 = ld_afrag(tile, 128, 0, lane);
  v16h a1 = ld_afrag(tile, 128, 32, lane);
  v16h a2 = ld_afrag(tile, 128, 64, lane);
  v16h a3 = ld_afrag(tile, 128, 96, lane);
  __syncthreads();

  // update layer 1: [16,128]@[128,64], ReLU -> stage stride-64
  for (int nt = 0; nt < 4; ++nt) {
    F8 acc; float b = bu1[nt * 16 + n0];
    for (int i = 0; i < 8; ++i) acc.f[i] = b;
    acc.v = wmma32(a0, ld_bfrag(wu1, 0 * 4 + nt, lane), acc.v);
    acc.v = wmma32(a1, ld_bfrag(wu1, 1 * 4 + nt, lane), acc.v);
    acc.v = wmma32(a2, ld_bfrag(wu1, 2 * 4 + nt, lane), acc.v);
    acc.v = wmma32(a3, ld_bfrag(wu1, 3 * 4 + nt, lane), acc.v);
    for (int i = 0; i < 8; ++i) {
      float v = acc.f[i]; v = v > 0.0f ? v : 0.0f;
      int m = (lane < 16) ? i : i + 8;
      tile[m * 64 + nt * 16 + n0] = (_Float16)v;
    }
  }
  __syncthreads();

  // update layer 2: [16,64]@[64,32] -> h_out, also stage f16 at tile+1024
  v16h u0 = ld_afrag(tile, 64, 0, lane);
  v16h u1 = ld_afrag(tile, 64, 32, lane);
  for (int nt = 0; nt < 2; ++nt) {
    F8 o; float b = bu2[nt * 16 + n0];
    for (int i = 0; i < 8; ++i) o.f[i] = b;
    o.v = wmma32(u0, ld_bfrag(wu2, 0 * 2 + nt, lane), o.v);
    o.v = wmma32(u1, ld_bfrag(wu2, 1 * 2 + nt, lane), o.v);
    for (int i = 0; i < 8; ++i) {
      int m = (lane < 16) ? i : i + 8;
      int nd = nb + m;
      if (nd < N) out[(size_t)nd * OUTD + nt * 16 + n0] = o.f[i];
      tile[1024 + m * 32 + nt * 16 + n0] = (_Float16)o.f[i];
    }
  }
  __syncthreads();

  // tactical layer 1: [16,32]@[32,64], ReLU -> stage stride-64 (region 0..1023)
  v16h ta = ld_afrag(tile + 1024, 32, 0, lane);
  for (int nt = 0; nt < 4; ++nt) {
    F8 acc; float b = bt1[nt * 16 + n0];
    for (int i = 0; i < 8; ++i) acc.f[i] = b;
    acc.v = wmma32(ta, ld_bfrag(wt1, nt, lane), acc.v);
    for (int i = 0; i < 8; ++i) {
      float v = acc.f[i]; v = v > 0.0f ? v : 0.0f;
      int m = (lane < 16) ? i : i + 8;
      tile[m * 64 + nt * 16 + n0] = (_Float16)v;
    }
  }
  __syncthreads();

  // tactical layer 2: [16,64]@[64,16], ReLU -> fp32 stage
  v16h s0 = ld_afrag(tile, 64, 0, lane);
  v16h s1 = ld_afrag(tile, 64, 32, lane);
  {
    F8 o; float b = bt2[n0];
    for (int i = 0; i < 8; ++i) o.f[i] = b;
    o.v = wmma32(s0, ld_bfrag(wt2, 0, lane), o.v);
    o.v = wmma32(s1, ld_bfrag(wt2, 1, lane), o.v);
    for (int i = 0; i < 8; ++i) {
      float v = o.f[i]; v = v > 0.0f ? v : 0.0f;
      int m = (lane < 16) ? i : i + 8;
      t2s[wave][m * 16 + n0] = v;
    }
  }
  __syncthreads();

  // tactical layer 3: [16,16]@[16,4] scalar (2 outputs per lane)
  {
    int cpair = (lane >> 4) * 2;
    int nd = nb + r;
    for (int cc = 0; cc < 2; ++cc) {
      int c = cpair + cc;
      float accv = bt3[c];
      for (int k = 0; k < 16; ++k)
        accv += t2s[wave][r * 16 + k] * Wt3[k * 4 + c];
      if (nd < N) out[(size_t)N * OUTD + (size_t)nd * 4 + c] = accv;
    }
  }
}

// ---------------- host launcher ----------------
extern "C" void kernel_launch(void* const* d_in, const int* in_sizes, int n_in,
                              void* d_out, int out_size, void* d_ws, size_t ws_size,
                              hipStream_t stream) {
  const float* feat   = (const float*)d_in[0];
  const int*   eidx   = (const int*)d_in[1];
  const float* encW1  = (const float*)d_in[2];
  const float* encb1  = (const float*)d_in[3];
  const float* encW2  = (const float*)d_in[4];
  const float* encb2  = (const float*)d_in[5];
  const float* msgW1  = (const float*)d_in[6];
  const float* msgb1  = (const float*)d_in[7];
  const float* msgW2  = (const float*)d_in[8];
  const float* msgb2  = (const float*)d_in[9];
  const float* updW1  = (const float*)d_in[10];
  const float* updb1  = (const float*)d_in[11];
  const float* updW2  = (const float*)d_in[12];
  const float* updb2  = (const float*)d_in[13];
  const float* tacW1  = (const float*)d_in[14];
  const float* tacb1  = (const float*)d_in[15];
  const float* tacW2  = (const float*)d_in[16];
  const float* tacb2  = (const float*)d_in[17];
  const float* tacW3  = (const float*)d_in[18];
  const float* tacb3  = (const float*)d_in[19];

  const int* erow = eidx;
  const int* ecol = eidx + NE;

  char* ws = (char*)d_ws;
  _Float16* h16    = (_Float16*)(ws + WS_H16);
  float*    sums   = (float*)(ws + WS_SUMS);
  float*    counts = (float*)(ws + WS_CNT);
  _Float16* pw     = (_Float16*)(ws + WS_W);

  _Float16* p_enc1 = pw + (size_t)F_ENC1 * 512;
  _Float16* p_enc2 = pw + (size_t)F_ENC2 * 512;
  _Float16* p_msg1 = pw + (size_t)F_MSG1 * 512;
  _Float16* p_msg2 = pw + (size_t)F_MSG2 * 512;
  _Float16* p_upd1 = pw + (size_t)F_UPD1 * 512;
  _Float16* p_upd2 = pw + (size_t)F_UPD2 * 512;
  _Float16* p_tac1 = pw + (size_t)F_TAC1 * 512;
  _Float16* p_tac2 = pw + (size_t)F_TAC2 * 512;

  float* outp = (float*)d_out;

  // 1) zero accumulators (sums immediately followed by counts in ws)
  long long zn = (long long)NN * HID + NN;
  gnn_zero_kernel<<<(unsigned)((zn + 255) / 256), 256, 0, stream>>>(sums, zn);

  // 2) pack all weights into WMMA B-fragment layout
  gnn_pack_weights<<<4,  512, 0, stream>>>(encW1, IN_DIM, HID, 4, p_enc1);
  gnn_pack_weights<<<8,  512, 0, stream>>>(encW2, HID,    HID, 4, p_enc2);
  gnn_pack_weights<<<16, 512, 0, stream>>>(msgW1, 2*HID,  HID, 4, p_msg1);
  gnn_pack_weights<<<8,  512, 0, stream>>>(msgW2, HID,    HID, 4, p_msg2);
  gnn_pack_weights<<<16, 512, 0, stream>>>(updW1, 2*HID,  HID, 4, p_upd1);
  gnn_pack_weights<<<4,  512, 0, stream>>>(updW2, HID,    OUTD, 2, p_upd2);
  gnn_pack_weights<<<4,  512, 0, stream>>>(tacW1, OUTD,   HID, 4, p_tac1);
  gnn_pack_weights<<<2,  512, 0, stream>>>(tacW2, HID,    16,  1, p_tac2);

  // 3) node encoder
  gnn_encode_kernel<<<(NN + 127) / 128, 256, 0, stream>>>(
      feat, p_enc1, encb1, p_enc2, encb2, h16, NN);

  // 4) degree counts
  gnn_count_kernel<<<(NE + 255) / 256, 256, 0, stream>>>(ecol, counts, NE);

  // 5) edge messages + scatter-add
  gnn_edge_kernel<<<(NE + 127) / 128, 256, 0, stream>>>(
      erow, ecol, h16, p_msg1, msgb1, p_msg2, msgb2, sums, NE);

  // 6) update MLP + tactical head
  gnn_update_kernel<<<(NN + 127) / 128, 256, 0, stream>>>(
      h16, sums, counts,
      p_upd1, updb1, p_upd2, updb2,
      p_tac1, tacb1, p_tac2, tacb2,
      tacW3, tacb3, outp, NN);

  (void)in_sizes; (void)n_in; (void)out_size; (void)ws_size;
}
